// DialogueGCNModel_50328426774950
// MI455X (gfx1250) — matrix operand order repaired and checked by reading, not compiled, not run on hardware
//
#include <hip/hip_runtime.h>
#include <hip/hip_bf16.h>

#define TT 128
#define BB 256
#define DD 1024
#define HH 150
#define GG 600          // 4H
#define NP 608          // 4H padded to 38*16
#define KR 160          // H padded for recurrent K
#define K1P 320         // 2H padded for layer-1 input K
#define OH 300          // 2H

typedef unsigned short u16;
typedef __bf16 bf16x16 __attribute__((ext_vector_type(16)));
typedef float f32x8 __attribute__((ext_vector_type(8)));

union BF16x16 { uint4 u[2]; bf16x16 v; };

__device__ __forceinline__ u16 f2bf(float f) {
  unsigned u = __float_as_uint(f);
  unsigned r = u + 0x7FFFu + ((u >> 16) & 1u);
  return (u16)(r >> 16);
}
__device__ __forceinline__ float bf2f(u16 h) {
  return __uint_as_float(((unsigned)h) << 16);
}
__device__ __forceinline__ float sigm(float x) { return 1.0f / (1.0f + expf(-x)); }

// gfx1250 async global->LDS copy of one 16-byte chunk (tracked by ASYNCcnt)
__device__ __forceinline__ void async_copy16(unsigned lds_off, const u16* gptr) {
  asm volatile("global_load_async_to_lds_b128 %0, %1, off"
               :: "v"(lds_off), "v"((unsigned long long)(size_t)gptr)
               : "memory");
}

// ---------------- small utility kernels ----------------

__global__ void zero_kernel(unsigned* __restrict__ p, size_t n) {
  size_t i = (size_t)blockIdx.x * 256 + threadIdx.x;
  if (i < n) p[i] = 0u;
}

__global__ void spk_pos_kernel(const int* __restrict__ qmask,
                               int* __restrict__ spk, int* __restrict__ pos) {
  int b = threadIdx.x;
  if (b >= BB) return;
  int c0 = 0, c1 = 0;
  for (int t = 0; t < TT; ++t) {
    size_t tb = (size_t)t * BB + b;
    int sp = qmask[tb * 2 + 1];            // one-hot -> argmax == q[...,1]
    int p;
    if (sp) { p = c1; c1++; } else { p = c0; c0++; }
    spk[tb] = sp;
    pos[tb] = p;
  }
}

__global__ void scatter_kernel(const float* __restrict__ U,
                               const int* __restrict__ spk, const int* __restrict__ pos,
                               u16* __restrict__ xpb, size_t total) {
  size_t i = (size_t)blockIdx.x * 256 + threadIdx.x;
  if (i >= total) return;
  int d = (int)(i & (DD - 1));
  size_t tb = i >> 10;
  int b = (int)(tb % BB);
  int sp = spk[tb];
  int p  = pos[tb];
  size_t dst = ((size_t)p * (2 * BB) + (size_t)sp * BB + b) * DD + d;
  xpb[dst] = f2bf(U[i]);
}

__global__ void cvt_f32_bf16_kernel(const float* __restrict__ src, u16* __restrict__ dst, size_t n) {
  size_t i = (size_t)blockIdx.x * 256 + threadIdx.x;
  if (i < n) dst[i] = f2bf(src[i]);
}

__global__ void convert_w_kernel(const float* __restrict__ src, u16* __restrict__ dst,
                                 int N, int K, int Kpad, int total) {
  int i = blockIdx.x * 256 + threadIdx.x;
  if (i >= total) return;
  int n = i / Kpad, k = i % Kpad;
  float v = (n < N && k < K) ? src[(size_t)n * K + k] : 0.0f;
  dst[i] = f2bf(v);
}

__global__ void bias_pad_kernel(const float* __restrict__ src, float* __restrict__ dst) {
  int i = blockIdx.x * 256 + threadIdx.x;
  if (i < NP) dst[i] = (i < GG) ? src[i] : 0.0f;
}

__global__ void combine_kernel(const float* __restrict__ emo, const float* __restrict__ prO,
                               const int* __restrict__ spk, const int* __restrict__ pos,
                               const float* __restrict__ umask, float* __restrict__ out,
                               size_t total) {
  size_t i = (size_t)blockIdx.x * 256 + threadIdx.x;
  if (i >= total) return;
  int n = (int)(i % OH);
  size_t tb = i / OH;
  int b = (int)(tb % BB);
  int sp = spk[tb];
  int p  = pos[tb];
  size_t prow = (size_t)p * (2 * BB) + (size_t)sp * BB + b;
  out[i] = (emo[tb * OH + n] + prO[prow * OH + n]) * umask[tb];
}

// ---------------- WMMA GEMM: C[M,NP](bf16) = A[M,K](bf16) @ W[NP,K]^T + bias ----------------
// block = (32,8): 8 waves each compute 16(M) x 32(N); the 32xK weight panel is shared by all
// waves and streamed through LDS in double-buffered 32x64 chunks via async global->LDS DMA.
// grid = (NP/32, M/128).  K must be a multiple of 64 (1024 and 320 both are).

__global__ __launch_bounds__(256) void gemm_bf16(const u16* __restrict__ A,
                                                 const u16* __restrict__ W,
                                                 const float* __restrict__ bias,
                                                 u16* __restrict__ C,
                                                 int M, int K) {
  __shared__ __align__(16) u16 bsm[2][32 * 64];   // [buf][n*64 + k]

  const int lane = threadIdx.x;
  const int wave = threadIdx.y;
  const int tid  = wave * 32 + lane;
  const int m0 = blockIdx.y * 128 + wave * 16;
  const int n0 = blockIdx.x * 32;
  const int hi = lane >> 4;
  const int l16 = lane & 15;

  // async staging: thread tid copies 16B chunk tid of the 32x64 panel
  // (panel row = tid>>3, k-chunk = tid&7; LDS layout is linear => offset tid*16)
  const u16* wsrc = W + (size_t)(n0 + (tid >> 3)) * K + ((tid & 7) << 3);
  const unsigned ldsme[2] = {
    (unsigned)(size_t)&bsm[0][0] + (unsigned)tid * 16u,
    (unsigned)(size_t)&bsm[1][0] + (unsigned)tid * 16u
  };

  f32x8 acc0, acc1;
  {
    float b0v = bias[n0 + l16];
    float b1v = bias[n0 + 16 + l16];
    for (int r = 0; r < 8; ++r) { acc0[r] = b0v; acc1[r] = b1v; }
  }

  const u16* arow = A + (size_t)(m0 + l16) * K + (hi << 3);
  const int nch = K >> 6;

  async_copy16(ldsme[0], wsrc);                       // prefetch chunk 0
  for (int c = 0; c < nch; ++c) {
    if (c + 1 < nch) {
      async_copy16(ldsme[(c + 1) & 1], wsrc + ((size_t)(c + 1) << 6));
      asm volatile("s_wait_asynccnt 0x1" ::: "memory");   // chunk c landed
    } else {
      asm volatile("s_wait_asynccnt 0x0" ::: "memory");
    }
    __syncthreads();

    const u16* bpan = &bsm[c & 1][0];
    const int kb = c << 6;
    for (int kk = 0; kk < 2; ++kk) {
      BF16x16 a, b0, b1;
      a.u[0] = *(const uint4*)(arow + kb + kk * 32);
      a.u[1] = *(const uint4*)(arow + kb + kk * 32 + 16);
      const u16* bp0 = bpan + l16 * 64 + kk * 32 + (hi << 3);
      const u16* bp1 = bpan + (16 + l16) * 64 + kk * 32 + (hi << 3);
      b0.u[0] = *(const uint4*)(bp0);
      b0.u[1] = *(const uint4*)(bp0 + 16);
      b1.u[0] = *(const uint4*)(bp1);
      b1.u[1] = *(const uint4*)(bp1 + 16);
      acc0 = __builtin_amdgcn_wmma_f32_16x16x32_bf16(false, a.v, false, b0.v, (short)0, acc0, false, false);
      acc1 = __builtin_amdgcn_wmma_f32_16x16x32_bf16(false, a.v, false, b1.v, (short)0, acc1, false, false);
    }
    __syncthreads();   // all reads of bsm[c&1] done before it is overwritten
  }

  u16* crow = C + (size_t)(m0 + hi * 8) * NP + n0 + l16;
  for (int r = 0; r < 8; ++r) {
    crow[(size_t)r * NP]      = f2bf(acc0[r]);
    crow[(size_t)r * NP + 16] = f2bf(acc1[r]);
  }
}

// ---------------- LSTM scan: one block per (16 batch rows, direction) ----------------
// Pre-gates for timestep t (16 x NP bf16, fully contiguous) are DMA'd into LDS with
// async copies; the matrix phase initializes WMMA accumulators from them and writes the
// recurrent result back in place; the gate phase applies the LSTM nonlinearity.

__global__ __launch_bounds__(256) void lstm_scan(const u16* __restrict__ gbuf,
                                                 const u16* __restrict__ whh,
                                                 u16* __restrict__ hbuf,
                                                 float* __restrict__ outF,
                                                 int Bn) {
  __shared__ __align__(16) u16 g_sm[16 * NP];   // staged pre-gates -> gate values (bf16)
  __shared__ __align__(16) u16 h_sm[16 * KR];   // hidden state, bf16, zero-padded cols
  __shared__ float c_sm[16 * HH];               // cell state, f32

  const int lane = threadIdx.x;
  const int wave = threadIdx.y;
  const int tid  = wave * 32 + lane;
  const int dir  = blockIdx.y;
  const int rowBase = blockIdx.x * 16;
  const int hi  = lane >> 4;
  const int l16 = lane & 15;

  const u16* gp = gbuf + (size_t)dir * TT * Bn * NP;
  const u16* wp = whh  + (size_t)dir * NP * KR;
  const unsigned glds = (unsigned)(size_t)&g_sm[0];

  for (int i = tid; i < 16 * KR; i += 256) h_sm[i] = 0;
  for (int i = tid; i < 16 * HH; i += 256) c_sm[i] = 0.0f;
  __syncthreads();

  for (int step = 0; step < TT; ++step) {
    const int t = dir ? (TT - 1 - step) : step;

    // ---- stage pre-gates for this timestep: 1216 x 16B async copies ----
    const u16* gsrc = gp + ((size_t)t * Bn + rowBase) * NP;
    for (int i = tid; i < (16 * NP) / 8; i += 256)
      async_copy16(glds + (unsigned)i * 16u, gsrc + (size_t)i * 8);
    asm volatile("s_wait_asynccnt 0x0" ::: "memory");
    __syncthreads();

    // ---- matrix phase: G = preGates + h @ Whh^T (in place in g_sm) ----
    BF16x16 afr[KR / 32];
    {
      const u16* hrow = h_sm + l16 * KR + (hi << 3);
      for (int kk = 0; kk < KR / 32; ++kk) {
        afr[kk].u[0] = *(const uint4*)(hrow + kk * 32);
        afr[kk].u[1] = *(const uint4*)(hrow + kk * 32 + 16);
      }
    }
    for (int j = wave; j < NP / 16; j += 8) {
      const int n = j * 16 + l16;
      f32x8 acc;
      for (int r = 0; r < 8; ++r)
        acc[r] = bf2f(g_sm[(hi * 8 + r) * NP + n]);
      const u16* wrow = wp + (size_t)n * KR + (hi << 3);
      for (int kk = 0; kk < KR / 32; ++kk) {
        BF16x16 b;
        b.u[0] = *(const uint4*)(wrow + kk * 32);
        b.u[1] = *(const uint4*)(wrow + kk * 32 + 16);
        acc = __builtin_amdgcn_wmma_f32_16x16x32_bf16(false, afr[kk].v, false, b.v, (short)0, acc, false, false);
      }
      for (int r = 0; r < 8; ++r)
        g_sm[(hi * 8 + r) * NP + n] = f2bf(acc[r]);   // same lane read these slots
    }
    __syncthreads();

    // ---- gate phase: i,f,g,o -> c,h ----
    for (int i = tid; i < 16 * HH; i += 256) {
      const int row = i / HH, n = i % HH;
      const float gi = bf2f(g_sm[row * NP + n]);
      const float gf = bf2f(g_sm[row * NP + HH + n]);
      const float gg = bf2f(g_sm[row * NP + 2 * HH + n]);
      const float go = bf2f(g_sm[row * NP + 3 * HH + n]);
      float c = sigm(gf) * c_sm[i] + sigm(gi) * tanhf(gg);
      float h = sigm(go) * tanhf(c);
      c_sm[i] = c;
      h_sm[row * KR + n] = f2bf(h);
      const size_t orow = (size_t)t * Bn + rowBase + row;
      hbuf[orow * K1P + dir * HH + n] = f2bf(h);
      if (outF) outF[orow * OH + dir * HH + n] = h;
    }
    __syncthreads();   // h/c/g consumed before next timestep overwrites
  }
}

// ---------------- host orchestration ----------------

extern "C" void kernel_launch(void* const* d_in, const int* in_sizes, int n_in,
                              void* d_out, int out_size, void* d_ws, size_t ws_size,
                              hipStream_t stream) {
  (void)in_sizes; (void)n_in; (void)out_size; (void)ws_size;

  const float* U      = (const float*)d_in[0];
  const int*   qmask  = (const int*)d_in[1];
  const float* umask  = (const float*)d_in[2];
  // d_in[3] seq_lengths: unused (all == T)
  const float* rW[2][3] = {
    { (const float*)d_in[4],  (const float*)d_in[5],  (const float*)d_in[6]  },   // rnn layer0
    { (const float*)d_in[10], (const float*)d_in[11], (const float*)d_in[12] } }; // pr layer0
  const float* rW1[2][3] = {
    { (const float*)d_in[7],  (const float*)d_in[8],  (const float*)d_in[9]  },   // rnn layer1
    { (const float*)d_in[13], (const float*)d_in[14], (const float*)d_in[15] } }; // pr layer1
  float* out = (float*)d_out;

  const size_t M_pr  = (size_t)TT * 2 * BB;   // 65536
  const size_t M_rnn = (size_t)TT * BB;       // 32768

  size_t off = 0;
  auto alloc = [&](size_t bytes) -> void* {
    void* p = (char*)d_ws + off;
    off = (off + bytes + 255) & ~(size_t)255;
    return p;
  };
  int*   spk    = (int*)  alloc((size_t)TT * BB * 4);
  int*   pos    = (int*)  alloc((size_t)TT * BB * 4);
  u16*   xpb    = (u16*)  alloc(M_pr  * DD * 2);
  u16*   ub     = (u16*)  alloc(M_rnn * DD * 2);
  u16*   gbuf   = (u16*)  alloc(2 * M_pr * NP * 2);
  u16*   hbuf   = (u16*)  alloc(M_pr * K1P * 2);
  float* prOut  = (float*)alloc(M_pr  * OH * 4);
  float* emoOut = (float*)alloc(M_rnn * OH * 4);
  u16* wih0b[2]; u16* whh0b[2]; u16* wih1b[2]; u16* whh1b[2];
  float* b0f[2]; float* b1f[2];
  for (int s = 0; s < 2; ++s) {
    wih0b[s] = (u16*)alloc((size_t)2 * NP * DD * 2);
    whh0b[s] = (u16*)alloc((size_t)2 * NP * KR * 2);
    b0f[s]   = (float*)alloc((size_t)2 * NP * 4);
    wih1b[s] = (u16*)alloc((size_t)2 * NP * K1P * 2);
    whh1b[s] = (u16*)alloc((size_t)2 * NP * KR * 2);
    b1f[s]   = (float*)alloc((size_t)2 * NP * 4);
  }

  auto zero = [&](void* p, size_t bytes) {
    size_t n = bytes / 4;
    zero_kernel<<<(unsigned)((n + 255) / 256), 256, 0, stream>>>((unsigned*)p, n);
  };
  auto cvt_w = [&](const float* src, u16* dst, int N, int K, int Kpad) {
    int total = NP * Kpad;
    for (int d = 0; d < 2; ++d)
      convert_w_kernel<<<(total + 255) / 256, 256, 0, stream>>>(
          src + (size_t)d * N * K, dst + (size_t)d * NP * Kpad, N, K, Kpad, total);
  };
  auto cvt_b = [&](const float* src, float* dst) {
    for (int d = 0; d < 2; ++d)
      bias_pad_kernel<<<(NP + 255) / 256, 256, 0, stream>>>(src + (size_t)d * GG, dst + (size_t)d * NP);
  };
  auto gemm = [&](const u16* A, const u16* Wt, const float* bias, u16* Cg, size_t M, int K) {
    dim3 g(NP / 32, (unsigned)(M / 128)), b(32, 8);
    gemm_bf16<<<g, b, 0, stream>>>(A, Wt, bias, Cg, (int)M, K);
  };
  auto scan = [&](const u16* gb, const u16* wh, u16* hb, float* of, int Bn) {
    lstm_scan<<<dim3(Bn / 16, 2), dim3(32, 8), 0, stream>>>(gb, wh, hb, of, Bn);
  };

  // ---- 1. speaker/pos + party scatter (bf16) ----
  spk_pos_kernel<<<1, 256, 0, stream>>>(qmask, spk, pos);
  zero(xpb, M_pr * DD * 2);
  {
    size_t total = M_rnn * DD;
    scatter_kernel<<<(unsigned)((total + 255) / 256), 256, 0, stream>>>(U, spk, pos, xpb, total);
    cvt_f32_bf16_kernel<<<(unsigned)((total + 255) / 256), 256, 0, stream>>>(U, ub, total);
  }

  // ---- 2. convert weights (s=0 rnn, s=1 pr) ----
  for (int s = 0; s < 2; ++s) {
    cvt_w(rW[s][0],  wih0b[s], GG, DD, DD);
    cvt_w(rW[s][1],  whh0b[s], GG, HH, KR);
    cvt_b(rW[s][2],  b0f[s]);
    cvt_w(rW1[s][0], wih1b[s], GG, OH, K1P);
    cvt_w(rW1[s][1], whh1b[s], GG, HH, KR);
    cvt_b(rW1[s][2], b1f[s]);
  }

  // ---- 3. pr stack (party sequences, Bn = 512) ----
  for (int d = 0; d < 2; ++d)
    gemm(xpb, wih0b[1] + (size_t)d * NP * DD, b0f[1] + (size_t)d * NP,
         gbuf + (size_t)d * M_pr * NP, M_pr, DD);
  zero(hbuf, M_pr * K1P * 2);
  scan(gbuf, whh0b[1], hbuf, nullptr, 2 * BB);
  for (int d = 0; d < 2; ++d)
    gemm(hbuf, wih1b[1] + (size_t)d * NP * K1P, b1f[1] + (size_t)d * NP,
         gbuf + (size_t)d * M_pr * NP, M_pr, K1P);
  scan(gbuf, whh1b[1], hbuf, prOut, 2 * BB);

  // ---- 4. rnn stack (dialogue order, Bn = 256) ----
  for (int d = 0; d < 2; ++d)
    gemm(ub, wih0b[0] + (size_t)d * NP * DD, b0f[0] + (size_t)d * NP,
         gbuf + (size_t)d * M_rnn * NP, M_rnn, DD);
  zero(hbuf, M_rnn * K1P * 2);
  scan(gbuf, whh0b[0], hbuf, nullptr, BB);
  for (int d = 0; d < 2; ++d)
    gemm(hbuf, wih1b[0] + (size_t)d * NP * K1P, b1f[0] + (size_t)d * NP,
         gbuf + (size_t)d * M_rnn * NP, M_rnn, K1P);
  scan(gbuf, whh1b[0], hbuf, emoOut, BB);

  // ---- 5. combine ----
  {
    size_t total = M_rnn * OH;
    combine_kernel<<<(unsigned)((total + 255) / 256), 256, 0, stream>>>(
        emoOut, prOut, spk, pos, umask, out, total);
  }
}
